// GoldenMoELayer_36885179138137
// MI455X (gfx1250) — matrix operand order repaired and checked by reading, not compiled, not run on hardware
//
#include <hip/hip_runtime.h>
#include <hip/hip_bf16.h>
#include <math.h>

// ---------------------------------------------------------------------------
// GoldenMoE dense-MoE forward for gfx1250 (MI455X).
// B=4, T=2048 (M=8192 tokens), D=1024, E=8, I=1024.
// bf16 WMMA (v_wmma_f32_16x16x32_bf16), fp32 accumulate.
// X tile staged into LDS via GLOBAL_LOAD_ASYNC_TO_LDS_B128 (ASYNCcnt path).
// Weights pre-transposed to bf16 in ws (48MB -> resident in 192MB L2).
// ---------------------------------------------------------------------------

#define DIM_D 1024
#define DIM_I 1024
#define NEXP  8
#define MTOK  8192           // B*T
#define MT    32             // token rows per workgroup in expert kernel

#define GOLDEN_CENTER 0.36787944117144233f   // 1/e
#define GOLDEN_LOWER  0.21231792754821912f   // 0.5 - ln(4/3)
#define GOLDEN_UPPER  0.5f

typedef __attribute__((ext_vector_type(16))) __bf16 bf16x16;
typedef __attribute__((ext_vector_type(8)))  float  v8f;

union Frag16 {              // one WMMA 16-bit operand: 16 bf16 per lane
    bf16x16 v;
    uint4   q[2];
};

__device__ __forceinline__ unsigned short f2bf(float f) {
    unsigned int u = __float_as_uint(f);
    u += 0x7FFFu + ((u >> 16) & 1u);       // round to nearest even
    return (unsigned short)(u >> 16);
}

__device__ __forceinline__ v8f v8zero() {
    v8f z;
#pragma unroll
    for (int i = 0; i < 8; ++i) z[i] = 0.0f;
    return z;
}

// Flat aperture rule (ISA 10.2): for LDS-space generic addresses,
// the LDS address is the low 32 bits; truncation yields the LDS byte offset.
__device__ __forceinline__ unsigned lds_off(const void* p) {
    return (unsigned)(uintptr_t)p;
}

// Async DMA: 16 bytes per lane, global -> LDS, no VGPR data path.
// Tracked by ASYNCcnt (s_wait_asynccnt).
__device__ __forceinline__ void async_copy_b128(unsigned lds_byte_off,
                                                const void* gsrc) {
    asm volatile("global_load_async_to_lds_b128 %0, %1, off"
                 :: "v"(lds_byte_off), "v"(gsrc) : "memory");
}

__device__ __forceinline__ void wait_async_zero() {
    asm volatile("s_wait_asynccnt 0x0" ::: "memory");
}

// A fragment (16x32 bf16): row-major source, row length `rowlen`.
// lane m = lane&15; half h = lane>>4.
//   elements[0..7]  = K[k0 + 8h .. +7]         (contiguous 16B)
//   elements[8..15] = K[k0 + 16 + 8h .. +7]    (contiguous 16B)
__device__ __forceinline__ void loadA(Frag16& f, const unsigned short* base,
                                      int row0, int k0, int rowlen, int lane) {
    const int m = row0 + (lane & 15);
    const int h = (lane >> 4) * 8;
    const unsigned short* p = base + (size_t)m * rowlen + k0 + h;
    f.q[0] = *(const uint4*)(p);
    f.q[1] = *(const uint4*)(p + 16);
}

// B fragment (32x16 bf16, KxN) from an N-major (transposed) matrix.
// lane n = n0 + (lane&15); half h = lane>>4.
//   elements[0..7]  = K[k0 + 8h .. +7]         (contiguous 16B)
//   elements[8..15] = K[k0 + 16 + 8h .. +7]    (contiguous 16B)
__device__ __forceinline__ void loadB(Frag16& f, const unsigned short* base,
                                      int n0, int k0, int rowlen, int lane) {
    const int n = n0 + (lane & 15);
    const int h = (lane >> 4) * 8;
    const unsigned short* p = base + (size_t)n * rowlen + k0 + h;
    f.q[0] = *(const uint4*)(p);
    f.q[1] = *(const uint4*)(p + 16);
}

#define WMMA_BF16(A, B, C)                                                    \
    __builtin_amdgcn_wmma_f32_16x16x32_bf16(false, (A).v, false, (B).v,       \
                                            (short)0, (C), false, false)

// ---------------------------------------------------------------------------
// 1) Routing weights: one thread per token.
// ---------------------------------------------------------------------------
__global__ void route_kernel(const float* __restrict__ x,
                             const float* __restrict__ Wr,
                             const float* __restrict__ temperature,
                             float* __restrict__ w_out) {
    const int t = blockIdx.x * blockDim.x + threadIdx.x;
    if (t >= MTOK) return;

    float acc[NEXP];
#pragma unroll
    for (int e = 0; e < NEXP; ++e) acc[e] = 0.0f;

    const float* xr = x + (size_t)t * DIM_D;
    for (int d = 0; d < DIM_D; ++d) {
        const float xv = xr[d];
        const float4 a = ((const float4*)(Wr + d * NEXP))[0];
        const float4 b = ((const float4*)(Wr + d * NEXP))[1];
        acc[0] += xv * a.x; acc[1] += xv * a.y;
        acc[2] += xv * a.z; acc[3] += xv * a.w;
        acc[4] += xv * b.x; acc[5] += xv * b.y;
        acc[6] += xv * b.z; acc[7] += xv * b.w;
    }

    const float invT = 1.0f / temperature[0];
    float dist[NEXP], wv[NEXP];
    float wsum = 0.0f;
#pragma unroll
    for (int e = 0; e < NEXP; ++e) {
        const float inh = 1.0f / (1.0f + __expf(-acc[e] * invT));
        const float dd  = fabsf(inh - GOLDEN_CENTER);
        const bool zone = (inh >= GOLDEN_LOWER) && (inh <= GOLDEN_UPPER);
        dist[e] = dd;
        wv[e]   = zone ? __expf(-dd * 10.0f) : 0.0f;
        wsum   += wv[e];
    }

    if (wsum < 1e-8f) {
        float fb[NEXP];
#pragma unroll
        for (int e = 0; e < NEXP; ++e) fb[e] = __expf(-dist[e] * (1.0f / 0.3f));
        int i1 = 0, i2 = -1;
        float v1 = -1e30f, v2 = -1e30f;
#pragma unroll
        for (int e = 0; e < NEXP; ++e) {
            if (fb[e] > v1) { v2 = v1; i2 = i1; v1 = fb[e]; i1 = e; }
            else if (fb[e] > v2) { v2 = fb[e]; i2 = e; }
        }
        const float inv = 1.0f / fmaxf(v1 + v2, 1e-8f);
#pragma unroll
        for (int e = 0; e < NEXP; ++e)
            w_out[(size_t)t * NEXP + e] =
                (e == i1 ? v1 : (e == i2 ? v2 : 0.0f)) * inv;
    } else {
        const float inv = 1.0f / fmaxf(wsum, 1e-8f);
#pragma unroll
        for (int e = 0; e < NEXP; ++e) w_out[(size_t)t * NEXP + e] = wv[e] * inv;
    }
}

// ---------------------------------------------------------------------------
// 2) x fp32 -> bf16 (flat), 4 elements per thread.
// ---------------------------------------------------------------------------
__global__ void cvt_x_kernel(const float4* __restrict__ src,
                             uint2* __restrict__ dst, int n4) {
    const int i = blockIdx.x * blockDim.x + threadIdx.x;
    if (i >= n4) return;
    const float4 v = src[i];
    uint2 o;
    o.x = (unsigned)f2bf(v.x) | ((unsigned)f2bf(v.y) << 16);
    o.y = (unsigned)f2bf(v.z) | ((unsigned)f2bf(v.w) << 16);
    dst[i] = o;
}

// ---------------------------------------------------------------------------
// 3) Per-expert 1024x1024 transpose + fp32->bf16 (LDS-tiled, coalesced).
//    dst[e][r][c] = bf16(src[e][c][r])
// ---------------------------------------------------------------------------
__global__ void transpose_cvt_kernel(const float* __restrict__ src,
                                     unsigned short* __restrict__ dst) {
    __shared__ float tile[32][33];
    const int N = 1024;
    const int e = blockIdx.z;
    src += (size_t)e * N * N;
    dst += (size_t)e * N * N;
    const int x0 = blockIdx.x * 32, y0 = blockIdx.y * 32;
    const int tx = threadIdx.x, ty = threadIdx.y;   // block (32, 8)
#pragma unroll
    for (int j = 0; j < 32; j += 8)
        tile[ty + j][tx] = src[(size_t)(y0 + ty + j) * N + x0 + tx];
    __syncthreads();
#pragma unroll
    for (int j = 0; j < 32; j += 8)
        dst[(size_t)(x0 + ty + j) * N + y0 + tx] = f2bf(tile[tx][ty + j]);
}

// ---------------------------------------------------------------------------
// 4) Zero the output.
// ---------------------------------------------------------------------------
__global__ void zero_kernel(float4* __restrict__ p, int n4) {
    const int i = blockIdx.x * blockDim.x + threadIdx.x;
    if (i < n4) p[i] = make_float4(0.f, 0.f, 0.f, 0.f);
}

// ---------------------------------------------------------------------------
// 5) Fused expert: H = silu(X Wg) * (X Wu); out += w_e * (H Wd).
//    One workgroup = 32 tokens. Each of 8 waves owns a 32(M)x32(N) register
//    block: 2 A-frags x 2 B-frags -> 8 WMMAs/k-step (phase 1), no duplicate
//    B traffic across waves. X tile staged via async DMA to LDS.
// ---------------------------------------------------------------------------
__global__ void __launch_bounds__(256)
moe_expert_kernel(const unsigned short* __restrict__ xb,    // [MTOK][D] bf16
                  const unsigned short* __restrict__ WgT,   // [I][D] bf16
                  const unsigned short* __restrict__ WuT,   // [I][D] bf16
                  const unsigned short* __restrict__ WdT,   // [D][I] bf16
                  const float* __restrict__ wroute,         // [MTOK][E]
                  float* __restrict__ out,                  // [MTOK][D]
                  int expert) {
    __shared__ unsigned short Xs[MT * DIM_D];   // 64 KB
    __shared__ unsigned short Hs[MT * DIM_I];   // 64 KB

    const int m0   = blockIdx.x * MT;
    const int tid  = threadIdx.x;
    const int lane = tid & 31;
    const int wid  = tid >> 5;          // 0..7 -> this wave's 32-col strip

    // ---- stage X tile (64KB) into LDS with async DMA ----------------------
    {
        const unsigned short* src = xb + (size_t)m0 * DIM_D;
        const unsigned xsBase = lds_off(Xs);
        for (int i = tid; i < (MT * DIM_D) / 8; i += 256)
            async_copy_b128(xsBase + (unsigned)i * 16u, src + (size_t)i * 8);
        wait_async_zero();
    }
    __syncthreads();

    // ---- phase 1: H = silu(X Wg) * (X Wu), bf16 into LDS ------------------
    for (int pass = 0; pass < 4; ++pass) {
        const int ncol0 = pass * 256 + wid * 32;

        // keep next pass's weight strips flowing into the near caches
        if (pass < 3) {
            const int nn = (pass + 1) * 256 + wid * 32;
            __builtin_prefetch(WgT + (size_t)nn * DIM_D, 0, 1);
            __builtin_prefetch(WuT + (size_t)nn * DIM_D, 0, 1);
        }

        v8f ag[2][2], au[2][2];
#pragma unroll
        for (int s = 0; s < 2; ++s)
#pragma unroll
            for (int t = 0; t < 2; ++t) { ag[s][t] = v8zero(); au[s][t] = v8zero(); }

        for (int k0 = 0; k0 < DIM_D; k0 += 32) {
            Frag16 a0, a1;
            loadA(a0, Xs, 0,  k0, DIM_D, lane);
            loadA(a1, Xs, 16, k0, DIM_D, lane);
#pragma unroll
            for (int t = 0; t < 2; ++t) {
                Frag16 bg, bu;
                loadB(bg, WgT, ncol0 + t * 16, k0, DIM_D, lane);
                loadB(bu, WuT, ncol0 + t * 16, k0, DIM_D, lane);
                ag[0][t] = WMMA_BF16(a0, bg, ag[0][t]);
                ag[1][t] = WMMA_BF16(a1, bg, ag[1][t]);
                au[0][t] = WMMA_BF16(a0, bu, au[0][t]);
                au[1][t] = WMMA_BF16(a1, bu, au[1][t]);
            }
        }

        const int n  = lane & 15;
        const int h2 = lane >> 4;
#pragma unroll
        for (int s = 0; s < 2; ++s) {
#pragma unroll
            for (int t = 0; t < 2; ++t) {
#pragma unroll
                for (int r = 0; r < 8; ++r) {
                    const float g  = ag[s][t][r];
                    const float u  = au[s][t][r];
                    const float sg = g / (1.0f + __expf(-g));   // silu
                    const int   m  = s * 16 + r + 8 * h2;
                    Hs[(size_t)m * DIM_I + ncol0 + t * 16 + n] = f2bf(sg * u);
                }
            }
        }
    }
    __syncthreads();

    // ---- phase 2: out += w_e * (H Wd) -------------------------------------
    float wrow[2][8];
    {
        const int h2 = lane >> 4;
#pragma unroll
        for (int s = 0; s < 2; ++s)
#pragma unroll
            for (int r = 0; r < 8; ++r)
                wrow[s][r] =
                    wroute[(size_t)(m0 + s * 16 + r + 8 * h2) * NEXP + expert];
    }

    for (int pass = 0; pass < 4; ++pass) {
        const int ncol0 = pass * 256 + wid * 32;

        if (pass < 3) {
            const int nn = (pass + 1) * 256 + wid * 32;
            __builtin_prefetch(WdT + (size_t)nn * DIM_I, 0, 1);
        }

        v8f acc[2][2];
#pragma unroll
        for (int s = 0; s < 2; ++s)
#pragma unroll
            for (int t = 0; t < 2; ++t) acc[s][t] = v8zero();

        for (int k0 = 0; k0 < DIM_I; k0 += 32) {
            Frag16 a0, a1;
            loadA(a0, Hs, 0,  k0, DIM_I, lane);
            loadA(a1, Hs, 16, k0, DIM_I, lane);
#pragma unroll
            for (int t = 0; t < 2; ++t) {
                Frag16 b;
                loadB(b, WdT, ncol0 + t * 16, k0, DIM_I, lane);
                acc[0][t] = WMMA_BF16(a0, b, acc[0][t]);
                acc[1][t] = WMMA_BF16(a1, b, acc[1][t]);
            }
        }

        const int n  = lane & 15;
        const int h2 = lane >> 4;
#pragma unroll
        for (int s = 0; s < 2; ++s) {
#pragma unroll
            for (int t = 0; t < 2; ++t) {
#pragma unroll
                for (int r = 0; r < 8; ++r) {
                    const int m = m0 + s * 16 + r + 8 * h2;
                    const size_t o = (size_t)m * DIM_D + ncol0 + t * 16 + n;
                    out[o] += wrow[s][r] * acc[s][t][r];
                }
            }
        }
    }
}

// ---------------------------------------------------------------------------
// Launch
// ---------------------------------------------------------------------------
extern "C" void kernel_launch(void* const* d_in, const int* in_sizes, int n_in,
                              void* d_out, int out_size, void* d_ws, size_t ws_size,
                              hipStream_t stream) {
    const float* x    = (const float*)d_in[0];   // [4,2048,1024]
    const float* Wg   = (const float*)d_in[1];   // [8,1024,1024] (e,d,i)
    const float* Wu   = (const float*)d_in[2];   // [8,1024,1024] (e,d,i)
    const float* Wd   = (const float*)d_in[3];   // [8,1024,1024] (e,i,d)
    const float* Wr   = (const float*)d_in[4];   // [1024,8]
    const float* temp = (const float*)d_in[5];   // [1]
    float* out = (float*)d_out;                  // [4,2048,1024]

    // ws layout: routing weights | x_bf16 | WgT | WuT | WdT  (~64.25 MB)
    char* ws = (char*)d_ws;
    float* w_rt = (float*)ws;                                      // 256 KB
    unsigned short* xb  = (unsigned short*)(ws + (size_t)(256 << 10));
    unsigned short* WgT = xb  + (size_t)MTOK * DIM_D;              // 16 MB each
    unsigned short* WuT = WgT + (size_t)NEXP * DIM_I * DIM_D;
    unsigned short* WdT = WuT + (size_t)NEXP * DIM_I * DIM_D;

    // 1) routing weights
    route_kernel<<<MTOK / 256, 256, 0, stream>>>(x, Wr, temp, w_rt);

    // 2) x -> bf16
    {
        const int n4 = (MTOK * DIM_D) / 4;
        cvt_x_kernel<<<n4 / 256, 256, 0, stream>>>((const float4*)x, (uint2*)xb, n4);
    }

    // 3) weight transpose+convert: WgT/WuT = [e][i][d], WdT = [e][d][i]
    {
        dim3 grid(32, 32, NEXP), block(32, 8, 1);
        transpose_cvt_kernel<<<grid, block, 0, stream>>>(Wg, WgT);
        transpose_cvt_kernel<<<grid, block, 0, stream>>>(Wu, WuT);
        transpose_cvt_kernel<<<grid, block, 0, stream>>>(Wd, WdT);
    }

    // 4) zero out
    {
        const int n4 = (MTOK * DIM_D) / 4;
        zero_kernel<<<n4 / 256, 256, 0, stream>>>((float4*)out, n4);
    }

    // 5) experts, sequential launches (stream-ordered => race-free out +=)
    for (int e = 0; e < NEXP; ++e) {
        moe_expert_kernel<<<MTOK / MT, 256, 0, stream>>>(
            xb,
            WgT + (size_t)e * DIM_I * DIM_D,
            WuT + (size_t)e * DIM_I * DIM_D,
            WdT + (size_t)e * DIM_I * DIM_D,
            w_rt, out, e);
    }
}
